// STraTS_MultiTimeAttention_10685878633064
// MI455X (gfx1250) — matrix operand order repaired and checked by reading, not compiled, not run on hardware
//
#include <hip/hip_runtime.h>
#include <cmath>

typedef __attribute__((ext_vector_type(16))) _Float16 v16h;
typedef __attribute__((ext_vector_type(8)))  _Float16 v8h;
typedef __attribute__((ext_vector_type(8)))  float    v8f;
typedef __attribute__((ext_vector_type(4)))  float    v4f;

static constexpr int BB  = 32;
static constexpr int SS  = 1024;
static constexpr int DD  = 256;
static constexpr int HH  = 8;
static constexpr int DKK = 32;
static constexpr float MASKV = -1e-30f;

union U16 { v16h v; v8h h2[2]; _Float16 e[16]; };

static __device__ __forceinline__ v8f wmma_f16(const v16h& a, const v16h& b, const v8f& c) {
    return __builtin_amdgcn_wmma_f32_16x16x32_f16(false, a, false, b, (short)0, c, false, false);
}

// ---------------------------------------------------------------------------
// Kernel 0: transpose + f32->f16 convert weights:  W[h][d][dk] -> WT[h][dk][d]
// ---------------------------------------------------------------------------
__global__ void wprep_kernel(const float* __restrict__ Wq,
                             const float* __restrict__ Wk,
                             const float* __restrict__ Wv,
                             _Float16* __restrict__ wt) {
    int idx = blockIdx.x * blockDim.x + threadIdx.x;
    const int per = HH * DD * DKK;                 // 65536
    if (idx >= 3 * per) return;
    int which = idx / per;
    int o = idx % per;                             // o = (h*DKK + dk)*DD + d
    int d  = o % DD;
    int dk = (o / DD) % DKK;
    int h  = o / (DD * DKK);
    const float* W = (which == 0) ? Wq : ((which == 1) ? Wk : Wv);
    wt[idx] = (_Float16)W[(h * DD + d) * DKK + dk];
}

// ---------------------------------------------------------------------------
// Kernel 1: projections.  in(f32 [B][S][D]) x WT(f16 [H][DK][D]) ->
//   qkv=0: Q  f16 [B][H][S][DK]
//   qkv=1: K  f16 [B][H][S][DK]
//   qkv=2: VT f16 [B][H][DK][S]   (transposed for P*V B-fragments)
// One wave = 16 sequence rows.  M=16, K=32 per WMMA, N split in two halves.
// ---------------------------------------------------------------------------
__global__ __launch_bounds__(256) void proj_kernel(const float* __restrict__ q_in,
                                                   const float* __restrict__ k_in,
                                                   const float* __restrict__ v_in,
                                                   const _Float16* __restrict__ wt,
                                                   _Float16* __restrict__ Qh,
                                                   _Float16* __restrict__ Kh,
                                                   _Float16* __restrict__ VTh) {
    const int wave  = threadIdx.x >> 5;
    const int lane  = threadIdx.x & 31;
    const int lrow  = lane & 15;        // N column / A row within tile
    const int lhalf = lane >> 4;        // which K-half this lane carries
    const int chunks = SS / 128;        // 8
    int bidx = blockIdx.x;
    int qkv  = bidx / (BB * chunks);
    int rem  = bidx % (BB * chunks);
    int b    = rem / chunks;
    int sc   = rem % chunks;
    int m0   = sc * 128 + wave * 16;

    const float* in = (qkv == 0) ? q_in : ((qkv == 1) ? k_in : v_in);
    const float* arow = in + ((size_t)b * SS + m0 + lrow) * DD + lhalf * 16;
    const _Float16* wbase = wt + (size_t)qkv * HH * DKK * DD;

    for (int h = 0; h < HH; ++h) {
        v8f acc0 = {}, acc1 = {};
        const _Float16* wh = wbase + (size_t)h * DKK * DD;
        for (int k0 = 0; k0 < DD; k0 += 32) {
            // --- A fragment (16x32 f16): lane L<16 holds K 0-15 of row L,
            //     lane L+16 holds K 16-31 of row L
            U16 a;
            v4f f0 = *(const v4f*)(arow + k0);
            v4f f1 = *(const v4f*)(arow + k0 + 4);
            v4f f2 = *(const v4f*)(arow + k0 + 8);
            v4f f3 = *(const v4f*)(arow + k0 + 12);
#pragma unroll
            for (int i = 0; i < 4; ++i) {
                a.e[i]      = (_Float16)f0[i];
                a.e[4 + i]  = (_Float16)f1[i];
                a.e[8 + i]  = (_Float16)f2[i];
                a.e[12 + i] = (_Float16)f3[i];
            }
            // --- B fragments (32x16): lane = N + 16*(K>=16); per lane 16
            //     contiguous K-values at fixed N -> WT rows are contiguous.
            const _Float16* bp0 = wh + (size_t)(lrow) * DD + k0 + lhalf * 16;
            const _Float16* bp1 = wh + (size_t)(16 + lrow) * DD + k0 + lhalf * 16;
            U16 b0, b1;
            b0.h2[0] = *(const v8h*)bp0;  b0.h2[1] = *(const v8h*)(bp0 + 8);
            b1.h2[0] = *(const v8h*)bp1;  b1.h2[1] = *(const v8h*)(bp1 + 8);
            acc0 = wmma_f16(a.v, b0.v, acc0);
            acc1 = wmma_f16(a.v, b1.v, acc1);
        }
        // --- store (C/D layout: VGPR r -> row r + 8*lhalf, col lrow)
        if (qkv < 2) {
            _Float16* outp = ((qkv == 0) ? Qh : Kh) + (size_t)(b * HH + h) * SS * DKK;
#pragma unroll
            for (int r = 0; r < 8; ++r) {
                int row = m0 + r + 8 * lhalf;
                outp[(size_t)row * DKK + lrow]      = (_Float16)acc0[r];
                outp[(size_t)row * DKK + 16 + lrow] = (_Float16)acc1[r];
            }
        } else {
            _Float16* outp = VTh + (size_t)(b * HH + h) * DKK * SS;
#pragma unroll
            for (int r = 0; r < 8; ++r) {
                int row = m0 + r + 8 * lhalf;
                outp[(size_t)lrow * SS + row]        = (_Float16)acc0[r];
                outp[(size_t)(16 + lrow) * SS + row] = (_Float16)acc1[r];
            }
        }
    }
}

// ---------------------------------------------------------------------------
// Kernel 2: flash attention per (b,h).  One wave = 16 query rows.
// ---------------------------------------------------------------------------
__global__ __launch_bounds__(256) void attn_kernel(const _Float16* __restrict__ Qh,
                                                   const _Float16* __restrict__ Kh,
                                                   const _Float16* __restrict__ VTh,
                                                   const float* __restrict__ mask,
                                                   float* __restrict__ out) {
    __shared__ _Float16 lds[8][16 * 32];   // per-wave private P tile (1 KB each)
    const int wave  = threadIdx.x >> 5;
    const int lane  = threadIdx.x & 31;
    const int lrow  = lane & 15;
    const int lhalf = lane >> 4;
    const int chunks = SS / 128;
    int bidx = blockIdx.x;
    int b   = bidx / (HH * chunks);
    int rem = bidx % (HH * chunks);
    int h   = rem / chunks;
    int qc  = rem % chunks;
    int m0  = qc * 128 + wave * 16;
    size_t bh = (size_t)(b * HH + h);

    // Q A-fragment: whole head dim (K = DK = 32) in one fragment, loaded once.
    const _Float16* qp = Qh + (bh * SS + m0 + lrow) * DKK + lhalf * 16;
    U16 qa;
    qa.h2[0] = *(const v8h*)qp;
    qa.h2[1] = *(const v8h*)(qp + 8);

    const _Float16* kbase = Kh  + bh * SS * DKK;
    const _Float16* vbase = VTh + bh * DKK * SS;
    const float*    mrowp = mask + (size_t)b * SS;

    float mx[8], lsum[8];
    v8f acc0 = {}, acc1 = {};
#pragma unroll
    for (int r = 0; r < 8; ++r) { mx[r] = -INFINITY; lsum[r] = 0.f; }
    _Float16* plds = &lds[wave][0];

    for (int j0 = 0; j0 < SS; j0 += 32) {
        // K B-fragments for key columns j0..j0+15 and j0+16..j0+31
        const _Float16* kp0 = kbase + (size_t)(j0 + lrow) * DKK + lhalf * 16;
        const _Float16* kp1 = kp0 + 16 * DKK;
        U16 kb0, kb1;
        kb0.h2[0] = *(const v8h*)kp0;  kb0.h2[1] = *(const v8h*)(kp0 + 8);
        kb1.h2[0] = *(const v8h*)kp1;  kb1.h2[1] = *(const v8h*)(kp1 + 8);
        if (j0 + 32 < SS) {   // hint next K/V tiles toward the caches
            __builtin_prefetch(kp0 + 32 * DKK, 0, 1);
            __builtin_prefetch(vbase + (size_t)lrow * SS + j0 + 32, 0, 1);
        }
        v8f s0 = {}, s1 = {};
        s0 = wmma_f16(qa.v, kb0.v, s0);
        s1 = wmma_f16(qa.v, kb1.v, s1);

        float mv0 = mrowp[j0 + lrow];
        float mv1 = mrowp[j0 + 16 + lrow];

#pragma unroll
        for (int r = 0; r < 8; ++r) {
            // exact reference masking: m*s + (1-m)*(-1e-30)
            float a0 = mv0 * s0[r] + (1.f - mv0) * MASKV;
            float a1 = mv1 * s1[r] + (1.f - mv1) * MASKV;
            // row max over the 32 key columns (2 per lane) held across lanes 0..15 of group
            float t = fmaxf(a0, a1);
#pragma unroll
            for (int off = 1; off < 16; off <<= 1)
                t = fmaxf(t, __shfl_xor(t, off, 32));
            float mn  = fmaxf(mx[r], t);
            float scl = __expf(mx[r] - mn);
            float p0  = __expf(a0 - mn);
            float p1  = __expf(a1 - mn);
            float ps  = p0 + p1;
#pragma unroll
            for (int off = 1; off < 16; off <<= 1)
                ps += __shfl_xor(ps, off, 32);
            lsum[r] = lsum[r] * scl + ps;
            mx[r]   = mn;
            acc0[r] *= scl;
            acc1[r] *= scl;
            // stage P into LDS (row-major 16x32 f16) for D-layout -> A-layout
            int prow = r + 8 * lhalf;
            plds[prow * 32 + lrow]      = (_Float16)p0;
            plds[prow * 32 + 16 + lrow] = (_Float16)p1;
        }
        // P A-fragment (16x32) from LDS (per-wave region, DS ops in-order)
        const _Float16* lp = plds + lrow * 32 + lhalf * 16;
        U16 pa;
        pa.h2[0] = *(const v8h*)lp;
        pa.h2[1] = *(const v8h*)(lp + 8);
        // V B-fragments from VT: per lane 16 contiguous keys at fixed dk
        const _Float16* vp0 = vbase + (size_t)lrow * SS + j0 + lhalf * 16;
        const _Float16* vp1 = vp0 + 16 * SS;
        U16 vb0, vb1;
        vb0.h2[0] = *(const v8h*)vp0;  vb0.h2[1] = *(const v8h*)(vp0 + 8);
        vb1.h2[0] = *(const v8h*)vp1;  vb1.h2[1] = *(const v8h*)(vp1 + 8);
        acc0 = wmma_f16(pa.v, vb0.v, acc0);
        acc1 = wmma_f16(pa.v, vb1.v, acc1);
    }

    // epilogue: normalize and write out[b][s][h*DK + dk] (f32)
#pragma unroll
    for (int r = 0; r < 8; ++r) {
        float inv = 1.f / lsum[r];
        int row = m0 + r + 8 * lhalf;
        float* op = out + ((size_t)b * SS + row) * (HH * DKK) + h * DKK;
        op[lrow]      = acc0[r] * inv;
        op[16 + lrow] = acc1[r] * inv;
    }
}

// ---------------------------------------------------------------------------
extern "C" void kernel_launch(void* const* d_in, const int* in_sizes, int n_in,
                              void* d_out, int out_size, void* d_ws, size_t ws_size,
                              hipStream_t stream) {
    (void)in_sizes; (void)n_in; (void)out_size; (void)ws_size;
    const float* query = (const float*)d_in[0];
    const float* key_t = (const float*)d_in[1];
    const float* value = (const float*)d_in[2];
    const float* mask  = (const float*)d_in[3];
    const float* Wq    = (const float*)d_in[4];
    const float* Wk    = (const float*)d_in[5];
    const float* Wv    = (const float*)d_in[6];
    float* out = (float*)d_out;

    _Float16* ws  = (_Float16*)d_ws;
    _Float16* WT  = ws;                                        // 3*H*DK*D halfs
    _Float16* Qh  = WT + (size_t)3 * HH * DKK * DD;            // B*H*S*DK halfs
    _Float16* Kh  = Qh + (size_t)BB * HH * SS * DKK;
    _Float16* VTh = Kh + (size_t)BB * HH * SS * DKK;

    int wtot = 3 * HH * DD * DKK;                              // 196608
    wprep_kernel<<<(wtot + 255) / 256, 256, 0, stream>>>(Wq, Wk, Wv, WT);
    proj_kernel<<<3 * BB * (SS / 128), 256, 0, stream>>>(query, key_t, value, WT, Qh, Kh, VTh);
    attn_kernel<<<BB * HH * (SS / 128), 256, 0, stream>>>(Qh, Kh, VTh, mask, out);
}
